// GAMMA_65558380806152
// MI455X (gfx1250) — matrix operand order repaired and checked by reading, not compiled, not run on hardware
//
#include <hip/hip_runtime.h>
#include <hip/hip_bf16.h>
#include <math.h>

// ---------------- problem constants (from reference) ----------------
#define B_    512
#define NG_   100
#define NN_   102        // depot + 100 + depot2
#define D_    128
#define H_    8
#define DH_   16
#define FF_   512
#define A_    2
#define T_    40
#define P3_   384        // W_proj output width (gk|gv|lk)
#define NEG_  (-1000000000.0f)
#define MTOK_ (B_*NN_)   // 52224 tokens: divisible by 128

// GEMM block tiling: 256 threads = 8 waves; block tile 128(M) x 64(N), K-step 32.
// Wave grid 4(M) x 2(N); each wave owns a 32x32 output = 2x2 WMMA tiles.
#define BM_ 128
#define BN_ 64
#define BK_ 32

typedef unsigned short bfu;                 // raw bf16 bits

// ---------------- WMMA types ----------------
typedef __attribute__((ext_vector_type(16))) __bf16         v16bf;
typedef __attribute__((ext_vector_type(16))) unsigned short v16us;
typedef __attribute__((ext_vector_type(8)))  float          v8f;

__device__ __forceinline__ bfu f2bf(float f) {
  union { float f; unsigned int u; } x; x.f = f;
  unsigned int u = x.u;
  unsigned int r = u + 0x7FFFu + ((u >> 16) & 1u);   // round-to-nearest-even
  return (bfu)(r >> 16);
}

// =====================================================================
// BF16 WMMA GEMM on pre-converted bf16 operands.
//   C[M,N] = A[M,K] @ W[K,N] (+bias f32) (optional ReLU)
//   Cf (f32) and/or Cbf (bf16) outputs, either may be null.
// M%128==0, N%64==0, K%32==0 hold for every GEMM in this model.
// Weight tile (32x64 bf16 = 4KB) staged via global_load_async_to_lds_b128
// (ASYNCcnt-tracked); A fragments are raw dword loads (2 bf16 each).
// =====================================================================
__global__ void gemm_bf16_wmma(const bfu* __restrict__ A,
                               const bfu* __restrict__ W,
                               const float* __restrict__ bias,
                               float* __restrict__ Cf,
                               bfu* __restrict__ Cbf,
                               int M, int K, int N, int relu)
{
  __shared__ bfu Bs[BK_][BN_];                         // 4 KB bf16 weight tile

  const int tid    = threadIdx.x;
  const int lane   = tid & 31;
  const int waveId = tid >> 5;
  const int wm     = waveId & 3;                       // wave row (0..3)
  const int wn     = waveId >> 2;                      // wave col (0..1)

  const int nBlk = N / BN_;
  const int bm0  = (blockIdx.x / nBlk) * BM_;
  const int bn0  = (blockIdx.x % nBlk) * BN_;

  const int half  = lane >> 4;                         // K-half select
  const int aKoff = half * 8;                          // A 16x32 bf16 lane map
  const int bKoff = half * 8;                          // B 32x16 bf16 lane map
  const int l15   = lane & 15;

  const int row0 = bm0 + wm * 32 + l15;
  const int row1 = row0 + 16;
  const int c0   = wn * 32 + l15;
  const int c1   = c0 + 16;

  const bfu* Arow0 = A + (size_t)row0 * K;
  const bfu* Arow1 = A + (size_t)row1 * K;
  const unsigned BsBase = (unsigned)(uintptr_t)&Bs[0][0];   // LDS offset = addr[31:0]

  // staging assignment: 256 b128 chunks (8 bf16 each) = 32 rows x 64 cols
  const int sRow = tid >> 3;                           // 0..31
  const int sCol = (tid & 7) << 3;                     // 0,8,...,56
  const unsigned sLds = BsBase + (((sRow << 6) + sCol) << 1);

  v8f c00 = {}, c01 = {}, c10 = {}, c11 = {};

  for (int k0 = 0; k0 < K; k0 += BK_) {
    // ---- async-stage 32x64 bf16 weight tile into LDS (1 b128 per thread)
    {
      const bfu* gp = W + (size_t)(k0 + sRow) * N + bn0 + sCol;
      asm volatile("global_load_async_to_lds_b128 %0, %1, off"
                   :: "v"(sLds), "v"((unsigned long long)(uintptr_t)gp)
                   : "memory");
      asm volatile("s_wait_asynccnt 0x0" ::: "memory");
    }
    __syncthreads();

    if (k0 + BK_ < K) {                                // prefetch next A slab
      __builtin_prefetch(Arow0 + k0 + BK_, 0, 1);
      __builtin_prefetch(Arow1 + k0 + BK_, 0, 1);
    }

    // ---- build fragments (no conversion: operands already bf16)
    v16us au0, au1, bu0, bu1;
#pragma unroll
    for (int v = 0; v < 8; ++v) {
      const int ka = k0 + ((v < 4) ? (aKoff + 2 * v) : (16 + aKoff + 2 * (v - 4)));
      const unsigned a0w = *(const unsigned*)(Arow0 + ka);   // 2 bf16 packed
      const unsigned a1w = *(const unsigned*)(Arow1 + ka);
      au0[2 * v]     = (bfu)(a0w & 0xFFFFu);
      au0[2 * v + 1] = (bfu)(a0w >> 16);
      au1[2 * v]     = (bfu)(a1w & 0xFFFFu);
      au1[2 * v + 1] = (bfu)(a1w >> 16);

      const int kb = (v < 4) ? (bKoff + 2 * v) : (16 + bKoff + 2 * (v - 4));   // 0..31 within tile
      bu0[2 * v]     = Bs[kb][c0];
      bu0[2 * v + 1] = Bs[kb + 1][c0];
      bu1[2 * v]     = Bs[kb][c1];
      bu1[2 * v + 1] = Bs[kb + 1][c1];
    }
    v16bf a0 = __builtin_bit_cast(v16bf, au0);
    v16bf a1 = __builtin_bit_cast(v16bf, au1);
    v16bf b0 = __builtin_bit_cast(v16bf, bu0);
    v16bf b1 = __builtin_bit_cast(v16bf, bu1);

    c00 = __builtin_amdgcn_wmma_f32_16x16x32_bf16(false, a0, false, b0, (short)0, c00, false, false);
    c01 = __builtin_amdgcn_wmma_f32_16x16x32_bf16(false, a0, false, b1, (short)0, c01, false, false);
    c10 = __builtin_amdgcn_wmma_f32_16x16x32_bf16(false, a1, false, b0, (short)0, c10, false, false);
    c11 = __builtin_amdgcn_wmma_f32_16x16x32_bf16(false, a1, false, b1, (short)0, c11, false, false);

    __syncthreads();                                   // before next tile overwrite
  }

  // ---- epilogue: bias / relu / store (C/D layout per ISA 7.12.2)
  const int gcol0 = bn0 + c0;
  const int gcol1 = bn0 + c1;
  const float bv0 = bias ? bias[gcol0] : 0.0f;
  const float bv1 = bias ? bias[gcol1] : 0.0f;
  const int rb0 = bm0 + wm * 32 + (half << 3);
  const int rb1 = rb0 + 16;
#pragma unroll
  for (int r = 0; r < 8; ++r) {
    float v00 = c00[r] + bv0, v01 = c01[r] + bv1;
    float v10 = c10[r] + bv0, v11 = c11[r] + bv1;
    if (relu) {
      v00 = fmaxf(v00, 0.0f); v01 = fmaxf(v01, 0.0f);
      v10 = fmaxf(v10, 0.0f); v11 = fmaxf(v11, 0.0f);
    }
    if (Cf) {
      Cf[(size_t)(rb0 + r) * N + gcol0] = v00;
      Cf[(size_t)(rb0 + r) * N + gcol1] = v01;
      Cf[(size_t)(rb1 + r) * N + gcol0] = v10;
      Cf[(size_t)(rb1 + r) * N + gcol1] = v11;
    }
    if (Cbf) {
      Cbf[(size_t)(rb0 + r) * N + gcol0] = f2bf(v00);
      Cbf[(size_t)(rb0 + r) * N + gcol1] = f2bf(v01);
      Cbf[(size_t)(rb1 + r) * N + gcol0] = f2bf(v10);
      Cbf[(size_t)(rb1 + r) * N + gcol1] = f2bf(v11);
    }
  }
}

// f32 -> bf16 bulk conversion (weights, once per call)
__global__ void convert_bf16_kernel(const float* __restrict__ in, bfu* __restrict__ out,
                                    int n)
{
  int i = blockIdx.x * blockDim.x + threadIdx.x;
  if (i < n) out[i] = f2bf(in[i]);
}

// =====================================================================
// Embedding: writes f32 (for BN/decode) and bf16 (for GEMM consumption)
// =====================================================================
__global__ void embed_kernel(const float* __restrict__ depot,
                             const float* __restrict__ loc,
                             const float* __restrict__ prize,
                             const float* __restrict__ depot2,
                             const float* __restrict__ Wd, const float* __restrict__ bd,
                             const float* __restrict__ Wn, const float* __restrict__ bnod,
                             float* __restrict__ emb, bfu* __restrict__ embbf)
{
  int idx = blockIdx.x * blockDim.x + threadIdx.x;
  if (idx >= B_ * NN_ * D_) return;
  int d = idx % D_;
  int n = (idx / D_) % NN_;
  int b = idx / (D_ * NN_);
  float s;
  if (n == 0) {
    s = depot[b * 2] * Wd[d] + depot[b * 2 + 1] * Wd[D_ + d] + bd[d];
  } else if (n == NN_ - 1) {
    s = depot2[b * 2] * Wd[d] + depot2[b * 2 + 1] * Wd[D_ + d] + bd[d];
  } else {
    int g = n - 1;
    s = loc[(b * NG_ + g) * 2] * Wn[d] + loc[(b * NG_ + g) * 2 + 1] * Wn[D_ + d]
      + prize[b * NG_ + g] * Wn[2 * D_ + d] + bnod[d];
  }
  emb[idx] = s;
  embbf[idx] = f2bf(s);
}

// =====================================================================
// MHA core: one block per (b,h); K/V staged in LDS; online softmax.
// Output bf16 only (consumed solely by the Wo GEMM).
// =====================================================================
__global__ void attn_kernel(const float* __restrict__ q,
                            const float* __restrict__ k,
                            const float* __restrict__ v,
                            bfu* __restrict__ o)
{
  const int b = blockIdx.x / H_;
  const int h = blockIdx.x % H_;
  __shared__ float Ks[NN_][DH_];
  __shared__ float Vs[NN_][DH_];
  const int tid = threadIdx.x;
  for (int idx = tid; idx < NN_ * DH_; idx += blockDim.x) {
    int n = idx / DH_, d = idx % DH_;
    size_t g = ((size_t)b * NN_ + n) * D_ + h * DH_ + d;
    Ks[n][d] = k[g];
    Vs[n][d] = v[g];
  }
  __syncthreads();
  if (tid < NN_) {
    float qi[DH_], acc[DH_];
#pragma unroll
    for (int d = 0; d < DH_; ++d) {
      qi[d] = q[((size_t)b * NN_ + tid) * D_ + h * DH_ + d];
      acc[d] = 0.0f;
    }
    float m = -1e30f, l = 0.0f;
    for (int j = 0; j < NN_; ++j) {
      float s = 0.0f;
#pragma unroll
      for (int d = 0; d < DH_; ++d) s += qi[d] * Ks[j][d];
      s *= 0.25f;                                   // 1/sqrt(DH)
      float mn = fmaxf(m, s);
      float corr = __expf(m - mn);
      float p = __expf(s - mn);
      l = l * corr + p;
#pragma unroll
      for (int d = 0; d < DH_; ++d) acc[d] = acc[d] * corr + p * Vs[j][d];
      m = mn;
    }
    float inv = 1.0f / l;
#pragma unroll
    for (int d = 0; d < DH_; ++d)
      o[((size_t)b * NN_ + tid) * D_ + h * DH_ + d] = f2bf(acc[d] * inv);
  }
}

// =====================================================================
// x = BN(x + r) per channel; writes f32 x and bf16 copy for GEMMs.
// =====================================================================
__global__ void bn_residual_kernel(float* __restrict__ x, bfu* __restrict__ xbf,
                                   const float* __restrict__ r,
                                   const float* __restrict__ g, const float* __restrict__ bb,
                                   int M)
{
  const int d = blockIdx.x;
  __shared__ float red[256], red2[256];
  float s = 0.0f, s2 = 0.0f;
  for (int i = threadIdx.x; i < M; i += blockDim.x) {
    float val = x[(size_t)i * D_ + d] + r[(size_t)i * D_ + d];
    s += val; s2 += val * val;
  }
  red[threadIdx.x] = s; red2[threadIdx.x] = s2;
  __syncthreads();
  for (int off = 128; off > 0; off >>= 1) {
    if (threadIdx.x < off) {
      red[threadIdx.x]  += red[threadIdx.x + off];
      red2[threadIdx.x] += red2[threadIdx.x + off];
    }
    __syncthreads();
  }
  const float mean  = red[0] / (float)M;
  const float var   = red2[0] / (float)M - mean * mean;
  const float scale = g[d] * rsqrtf(var + 1e-5f);
  const float shift = bb[d];
  for (int i = threadIdx.x; i < M; i += blockDim.x) {
    float val = x[(size_t)i * D_ + d] + r[(size_t)i * D_ + d];
    float y = scale * (val - mean) + shift;
    x[(size_t)i * D_ + d]   = y;
    xbf[(size_t)i * D_ + d] = f2bf(y);
  }
}

// mean over N -> (B, D) bf16 (consumed only by W_fixed GEMM)
__global__ void mean_kernel(const float* __restrict__ emb, bfu* __restrict__ out)
{
  const int b = blockIdx.x, d = threadIdx.x;
  float s = 0.0f;
  for (int n = 0; n < NN_; ++n) s += emb[((size_t)b * NN_ + n) * D_ + d];
  out[b * D_ + d] = f2bf(s * (1.0f / (float)NN_));
}

// =====================================================================
// Decoder state init
// =====================================================================
__global__ void decode_init(unsigned int* __restrict__ visited, int* __restrict__ prev,
                            float* __restrict__ remaining, float* __restrict__ ll,
                            const float* __restrict__ max_length)
{
  int b = blockIdx.x * blockDim.x + threadIdx.x;
  if (b >= B_) return;
  visited[b * 4 + 0] = 1u;      // node 0 visited
  visited[b * 4 + 1] = 0u;
  visited[b * 4 + 2] = 0u;
  visited[b * 4 + 3] = 0u;
  for (int k = 0; k < A_; ++k) {
    prev[b * A_ + k] = 0;
    remaining[b * A_ + k] = max_length[b];
    ll[b * A_ + k] = 0.0f;
  }
}

__device__ __forceinline__ void coord_of(int n, int b,
                                         const float* depot, const float* loc,
                                         const float* depot2, float* c)
{
  if (n == 0)            { c[0] = depot[b * 2];  c[1] = depot[b * 2 + 1]; }
  else if (n == NN_ - 1) { c[0] = depot2[b * 2]; c[1] = depot2[b * 2 + 1]; }
  else {
    int g = n - 1;
    c[0] = loc[(b * NG_ + g) * 2];
    c[1] = loc[(b * NG_ + g) * 2 + 1];
  }
}

// =====================================================================
// One (t, k) decode step: block per batch element, 128 threads, LDS-staged.
// =====================================================================
__global__ void decode_step(const float* __restrict__ emb, const float* __restrict__ proj,
                            const float* __restrict__ fixedc,
                            const float* __restrict__ Wstep, const float* __restrict__ Wout,
                            const float* __restrict__ depot, const float* __restrict__ loc,
                            const float* __restrict__ depot2,
                            unsigned int* __restrict__ visited, int* __restrict__ prev,
                            float* __restrict__ remaining, float* __restrict__ llacc,
                            float* __restrict__ out_pi, int t, int k)
{
  const int b = blockIdx.x, tid = threadIdx.x;
  __shared__ float embp[D_], qv[D_], comp[H_][NN_], mrow[H_], lrow[H_];
  __shared__ float gl[D_], g2[D_], logits[NN_];
  __shared__ int   msk[NN_];

  const int pv = prev[b * A_ + k];
  const float rem = remaining[b * A_ + k];
  embp[tid] = emb[((size_t)b * NN_ + pv) * D_ + tid];
  __syncthreads();

  {
    float s = fixedc[b * D_ + tid];
    for (int j = 0; j < D_; ++j) s += embp[j] * Wstep[j * D_ + tid];
    s += rem * Wstep[D_ * D_ + tid];
    qv[tid] = s;
  }
  __syncthreads();

  if (tid < NN_) {
    int vis = (visited[b * 4 + (tid >> 5)] >> (tid & 31)) & 1;
    int mk = vis && (tid != NN_ - 1);
    msk[tid] = mk;
    const float* gK = proj + ((size_t)b * NN_ + tid) * P3_;     // gk cols [0,128)
#pragma unroll
    for (int h = 0; h < H_; ++h) {
      float s = 0.0f;
#pragma unroll
      for (int d = 0; d < DH_; ++d) s += qv[h * DH_ + d] * gK[h * DH_ + d];
      comp[h][tid] = mk ? NEG_ : s * 0.25f;
    }
  }
  __syncthreads();
  if (tid < H_) {
    float m = -1e30f;
    for (int n = 0; n < NN_; ++n) m = fmaxf(m, comp[tid][n]);
    float l = 0.0f;
    for (int n = 0; n < NN_; ++n) l += __expf(comp[tid][n] - m);
    mrow[tid] = m; lrow[tid] = l;
  }
  __syncthreads();
  if (tid < NN_) {
#pragma unroll
    for (int h = 0; h < H_; ++h)
      comp[h][tid] = __expf(comp[h][tid] - mrow[h]) / lrow[h];
  }
  __syncthreads();

  {
    const int h = tid >> 4, dh = tid & 15;
    float s = 0.0f;
    for (int n = 0; n < NN_; ++n)
      s += comp[h][n] * proj[((size_t)b * NN_ + n) * P3_ + D_ + h * DH_ + dh];
    gl[tid] = s;
  }
  __syncthreads();
  {
    float s = 0.0f;
    for (int j = 0; j < D_; ++j) s += gl[j] * Wout[j * D_ + tid];
    g2[tid] = s;
  }
  __syncthreads();
  if (tid < NN_) {
    const float* lk = proj + ((size_t)b * NN_ + tid) * P3_ + 2 * D_;
    float s = 0.0f;
    for (int d = 0; d < D_; ++d) s += g2[d] * lk[d];
    float val = 10.0f * tanhf(s * 0.08838834764831845f);  // CLIP * tanh(./sqrt(128))
    logits[tid] = msk[tid] ? NEG_ : val;
  }
  __syncthreads();

  if (tid == 0) {
    float mx = -1e30f; int sel = 0;
    for (int n = 0; n < NN_; ++n)
      if (logits[n] > mx) { mx = logits[n]; sel = n; }
    float se = 0.0f;
    for (int n = 0; n < NN_; ++n) se += __expf(logits[n] - mx);
    float lp = logits[sel] - mx - __logf(se);
    llacc[b * A_ + k] += lp;

    float cs[2], cp[2];
    coord_of(sel, b, depot, loc, depot2, cs);
    coord_of(pv,  b, depot, loc, depot2, cp);
    float dx = cs[0] - cp[0], dy = cs[1] - cp[1];
    remaining[b * A_ + k] = rem - sqrtf(dx * dx + dy * dy);
    visited[b * 4 + (sel >> 5)] |= (1u << (sel & 31));
    prev[b * A_ + k] = sel;
    out_pi[((size_t)b * T_ + t) * A_ + k] = (float)sel;
  }
}

__global__ void write_ll(const float* __restrict__ llacc, float* __restrict__ out)
{
  int i = blockIdx.x * blockDim.x + threadIdx.x;
  if (i < B_ * A_) out[i] = llacc[i];
}

// =====================================================================
// Host side
// =====================================================================
static inline void launch_gemm(const bfu* A, const bfu* W, const float* bias,
                               float* Cf, bfu* Cbf, int M, int K, int N, int relu,
                               hipStream_t stream)
{
  int blocks = (M / BM_) * (N / BN_);
  gemm_bf16_wmma<<<blocks, 256, 0, stream>>>(A, W, bias, Cf, Cbf, M, K, N, relu);
}

static inline void launch_cvt(const float* in, bfu* out, int n, hipStream_t stream)
{
  convert_bf16_kernel<<<(n + 255) / 256, 256, 0, stream>>>(in, out, n);
}

extern "C" void kernel_launch(void* const* d_in, const int* in_sizes, int n_in,
                              void* d_out, int out_size, void* d_ws, size_t ws_size,
                              hipStream_t stream)
{
  // ---- inputs (setup_inputs insertion order; params leaves in insertion order)
  const float* depot      = (const float*)d_in[0];
  const float* loc        = (const float*)d_in[1];
  const float* prize      = (const float*)d_in[2];
  const float* depot2     = (const float*)d_in[3];
  const float* max_length = (const float*)d_in[4];
  const float* W_depot = (const float*)d_in[5];
  const float* b_depot = (const float*)d_in[6];
  const float* W_node  = (const float*)d_in[7];
  const float* b_node  = (const float*)d_in[8];
  const int ENC0 = 9;                             // 12 leaves per encoder layer
  const float* W_proj  = (const float*)d_in[ENC0 + 2 * 12 + 0];
  const float* W_fixed = (const float*)d_in[ENC0 + 2 * 12 + 1];
  const float* W_step  = (const float*)d_in[ENC0 + 2 * 12 + 2];
  const float* W_out   = (const float*)d_in[ENC0 + 2 * 12 + 3];

  // ---- workspace layout (bytes), with region aliasing
  char* wsb = (char*)d_ws;
  const size_t SZ_X = (size_t)MTOK_ * D_;                  // 6,684,672 elements
  float* x    = (float*)wsb;                               // f32 activations
  bfu*   xbf  = (bfu*)(wsb + SZ_X * 4);                    // bf16 copy of x
  char*  r1b  = wsb + SZ_X * 4 + SZ_X * 2;                 // big aliased region (4*SZ_X f32)
  float* qb   = (float*)r1b;
  float* kb   = (float*)(r1b + SZ_X * 4);
  float* vb   = (float*)(r1b + 2 * SZ_X * 4);
  bfu*   obf  = (bfu*)(r1b + 3 * SZ_X * 4);                // attn out bf16
  bfu*   hbf  = (bfu*)r1b;                                 // FFN hidden bf16 (MTOK*512) aliases q..v
  float* projb= (float*)r1b;                               // proj f32 (MTOK*384) aliases after encoder
  char*  tail = r1b + 4 * SZ_X * 4;
  float* fb     = (float*)tail;                            // Wo-out / FFN-out f32
  float* meanf  = (float*)(tail + SZ_X * 4);               // not used as f32 output; spare
  bfu*   meanbf = (bfu*)meanf;                             // B*D bf16
  float* fixedc = (float*)(tail + SZ_X * 4 + (size_t)B_ * D_ * 4);
  bfu*   wpool  = (bfu*)(tail + SZ_X * 4 + 2 * (size_t)B_ * D_ * 4);
  // bf16 weight pool: 2 layers * (4*128*128 + 128*512 + 512*128) + 128*384 + 128*128
  bfu* Wq_bf[2], *Wk_bf[2], *Wv_bf[2], *Wo_bf[2], *W1_bf[2], *W2_bf[2];
  bfu* wp = wpool;
  for (int l = 0; l < 2; ++l) {
    Wq_bf[l] = wp; wp += D_ * D_;
    Wk_bf[l] = wp; wp += D_ * D_;
    Wv_bf[l] = wp; wp += D_ * D_;
    Wo_bf[l] = wp; wp += D_ * D_;
    W1_bf[l] = wp; wp += D_ * FF_;
    W2_bf[l] = wp; wp += FF_ * D_;
  }
  bfu* Wproj_bf  = wp; wp += D_ * P3_;
  bfu* Wfixed_bf = wp; wp += D_ * D_;
  float* state   = (float*)(((uintptr_t)wp + 255) & ~(uintptr_t)255);
  unsigned int* visited = (unsigned int*)state;            // B*4 u32
  int*   prev      = (int*)(state + 2048);                 // B*A
  float* remaining = state + 2048 + 1024;                  // B*A
  float* llacc     = state + 2048 + 2048;                  // B*A

  float* out_ll = (float*)d_out;                           // (B, A)
  float* out_pi = (float*)d_out + (size_t)B_ * A_;         // (B, T, A)

  // ---- weight pre-conversion to bf16 (tiny; once per call)
  for (int l = 0; l < 2; ++l) {
    const int base = ENC0 + l * 12;
    launch_cvt((const float*)d_in[base + 0], Wq_bf[l], D_ * D_, stream);
    launch_cvt((const float*)d_in[base + 1], Wk_bf[l], D_ * D_, stream);
    launch_cvt((const float*)d_in[base + 2], Wv_bf[l], D_ * D_, stream);
    launch_cvt((const float*)d_in[base + 3], Wo_bf[l], D_ * D_, stream);
    launch_cvt((const float*)d_in[base + 6], W1_bf[l], D_ * FF_, stream);
    launch_cvt((const float*)d_in[base + 8], W2_bf[l], FF_ * D_, stream);
  }
  launch_cvt(W_proj,  Wproj_bf,  D_ * P3_, stream);
  launch_cvt(W_fixed, Wfixed_bf, D_ * D_, stream);

  // ---- embedding (f32 + bf16)
  {
    int tot = B_ * NN_ * D_;
    embed_kernel<<<(tot + 255) / 256, 256, 0, stream>>>(
        depot, loc, prize, depot2, W_depot, b_depot, W_node, b_node, x, xbf);
  }

  // ---- encoder (2 layers)
  for (int l = 0; l < 2; ++l) {
    const int base = ENC0 + l * 12;
    const float* g1  = (const float*)d_in[base + 4];
    const float* b1  = (const float*)d_in[base + 5];
    const float* bf1 = (const float*)d_in[base + 7];
    const float* bf2 = (const float*)d_in[base + 9];
    const float* g2  = (const float*)d_in[base + 10];
    const float* b2  = (const float*)d_in[base + 11];

    launch_gemm(xbf, Wq_bf[l], nullptr, qb, nullptr, MTOK_, D_, D_, 0, stream);
    launch_gemm(xbf, Wk_bf[l], nullptr, kb, nullptr, MTOK_, D_, D_, 0, stream);
    launch_gemm(xbf, Wv_bf[l], nullptr, vb, nullptr, MTOK_, D_, D_, 0, stream);
    attn_kernel<<<B_ * H_, 128, 0, stream>>>(qb, kb, vb, obf);
    launch_gemm(obf, Wo_bf[l], nullptr, fb, nullptr, MTOK_, D_, D_, 0, stream);
    bn_residual_kernel<<<D_, 256, 0, stream>>>(x, xbf, fb, g1, b1, MTOK_);
    launch_gemm(xbf, W1_bf[l], bf1, nullptr, hbf, MTOK_, D_, FF_, 1, stream);  // bf16-only out
    launch_gemm(hbf, W2_bf[l], bf2, fb, nullptr, MTOK_, FF_, D_, 0, stream);
    bn_residual_kernel<<<D_, 256, 0, stream>>>(x, xbf, fb, g2, b2, MTOK_);
  }

  // ---- decoder precompute
  mean_kernel<<<B_, 128, 0, stream>>>(x, meanbf);
  launch_gemm(meanbf, Wfixed_bf, nullptr, fixedc, nullptr, B_, D_, D_, 0, stream);
  launch_gemm(xbf, Wproj_bf, nullptr, projb, nullptr, MTOK_, D_, P3_, 0, stream);

  // ---- sequential decode: T steps x A agents
  decode_init<<<(B_ + 255) / 256, 256, 0, stream>>>(visited, prev, remaining, llacc,
                                                    max_length);
  for (int t = 0; t < T_; ++t)
    for (int k = 0; k < A_; ++k)
      decode_step<<<B_, 128, 0, stream>>>(x, projb, fixedc, W_step, W_out,
                                          depot, loc, depot2,
                                          visited, prev, remaining, llacc,
                                          out_pi, t, k);

  write_ll<<<(B_ * A_ + 255) / 256, 256, 0, stream>>>(llacc, out_ll);
}
